// CausalSelfAttention_18056042513126
// MI455X (gfx1250) — matrix operand order repaired and checked by reading, not compiled, not run on hardware
//
#include <hip/hip_runtime.h>

// ---------------------------------------------------------------------------
// CDNA5 (gfx1250) causal self-attention, bf16 WMMA pipeline, wave32.
// GEMMs use 64x64 register-blocked macro-tiles (4x4 grid of 16x16 WMMA tiles)
// for 4x arithmetic-intensity vs a naive per-tile wave.
// ---------------------------------------------------------------------------

typedef __attribute__((ext_vector_type(16))) __bf16 v16bf;
typedef __attribute__((ext_vector_type(8)))  float  v8f;

struct Frag {
  union { v16bf v; unsigned int d[8]; };
};

__device__ __forceinline__ unsigned short f32_to_bf16(float f) {
  unsigned int u = __float_as_uint(f);
  u += 0x7FFFu + ((u >> 16) & 1u);          // round-to-nearest-even
  return (unsigned short)(u >> 16);
}

// A-matrix (16x32 bf16) per-lane K-pair start: ISA 7.12.2 16-bit A layout.
__device__ __forceinline__ int a_kpair(int v, int hi) {
  return (v < 4) ? (hi * 8 + 2 * v) : (16 + hi * 8 + 2 * (v - 4));
}
// B-matrix (32x16 bf16) per-lane K-pair start.
__device__ __forceinline__ int b_kpair(int v, int hi) {
  return (v < 4) ? (hi * 8 + 2 * v) : (16 + hi * 8 + 2 * (v - 4));
}

__device__ __forceinline__ v8f wmma_bf16(const Frag& a, const Frag& b, v8f c) {
  return __builtin_amdgcn_wmma_f32_16x16x32_bf16(false, a.v, false, b.v,
                                                 (short)0, c, false, false);
}

// 64x64 macro-tile of C = A * B^T, K = 640, bf16 operands row-major along K.
// acc[mi][ni] is the 16x16 tile (mbase+16*mi, nbase+16*ni).
__device__ __forceinline__ void gemm_64x64_k640(const unsigned short* __restrict__ Abase,
                                                const unsigned short* __restrict__ Bbase,
                                                int mbase, int nbase, int lane,
                                                v8f acc[4][4]) {
  const int hi = lane >> 4, ln = lane & 15;
  const unsigned short* ar[4];
  const unsigned short* br[4];
#pragma unroll
  for (int i = 0; i < 4; ++i) {
    ar[i] = Abase + (size_t)(mbase + i * 16 + ln) * 640;
    br[i] = Bbase + (size_t)(nbase + i * 16 + ln) * 640;
  }
  for (int ks = 0; ks < 640; ks += 32) {
    Frag a[4], b[4];
#pragma unroll
    for (int i = 0; i < 4; ++i) {
#pragma unroll
      for (int v = 0; v < 8; ++v) {
        a[i].d[v] = *(const unsigned int*)(ar[i] + ks + a_kpair(v, hi));
        b[i].d[v] = *(const unsigned int*)(br[i] + ks + b_kpair(v, hi));
      }
    }
#pragma unroll
    for (int mi = 0; mi < 4; ++mi)
#pragma unroll
      for (int ni = 0; ni < 4; ++ni)
        acc[mi][ni] = wmma_bf16(a[mi], b[ni], acc[mi][ni]);
  }
}

// ---------------------------------------------------------------------------
// K0: fp32 -> bf16 conversion
// ---------------------------------------------------------------------------
__global__ __launch_bounds__(256)
void cvt_bf16_kernel(const float* __restrict__ in, unsigned short* __restrict__ out, int n) {
  int i = blockIdx.x * blockDim.x + threadIdx.x;
  if (i < n) out[i] = f32_to_bf16(in[i]);
}

// ---------------------------------------------------------------------------
// K1: qkv = x @ w_qkv^T ; scatter q,k -> [B,H,S,64], v -> [B,H,64,S] (bf16)
// 128 x 30 macro-tiles (64x64), one per wave.
// ---------------------------------------------------------------------------
__global__ __launch_bounds__(128)
void qkv_gemm_kernel(const unsigned short* __restrict__ xb,
                     const unsigned short* __restrict__ wb,
                     unsigned short* __restrict__ qbuf,
                     unsigned short* __restrict__ kbuf,
                     unsigned short* __restrict__ vbuf) {
  const int lane = threadIdx.x & 31;
  const int hi = lane >> 4, ln = lane & 15;
  const int tile = blockIdx.x * 4 + (threadIdx.x >> 5);
  const int mt = tile / 30, nt = tile % 30;
  const int mbase = mt * 64, nbase = nt * 64;

  v8f acc[4][4] = {};
  gemm_64x64_k640(xb, wb, mbase, nbase, lane, acc);

#pragma unroll
  for (int mi = 0; mi < 4; ++mi)
#pragma unroll
    for (int ni = 0; ni < 4; ++ni) {
      const int col = nbase + ni * 16 + ln;
      const int which = col / 640, e = col % 640;
      const int h = e >> 6, d = e & 63;
#pragma unroll
      for (int r = 0; r < 8; ++r) {
        const int token = mbase + mi * 16 + r + 8 * hi;
        const int b = token >> 11, s = token & 2047;
        const unsigned short val = f32_to_bf16(acc[mi][ni][r]);
        const size_t bh = (size_t)b * 10 + h;
        if (which == 0)      qbuf[(bh * 2048 + s) * 64 + d] = val;
        else if (which == 1) kbuf[(bh * 2048 + s) * 64 + d] = val;
        else                 vbuf[(bh * 64 + d) * 2048 + s] = val;   // transposed
      }
    }
}

// ---------------------------------------------------------------------------
// K2: flash attention. grid = (128 q-tiles, 10 bh-groups), 4 waves/block,
// wave w handles (bh = blockIdx.y*4 + w, q rows qbase..qbase+15).
// ---------------------------------------------------------------------------
__global__ __launch_bounds__(128)
void attn_kernel(const unsigned short* __restrict__ qbuf,
                 const unsigned short* __restrict__ kbuf,
                 const unsigned short* __restrict__ vbuf,
                 unsigned short* __restrict__ yb) {
  __shared__ unsigned short pbuf[4][16][32];   // per-wave P tile (bf16)

  const int lane = threadIdx.x & 31;
  const int widx = threadIdx.x >> 5;
  const int hi = lane >> 4, ln = lane & 15;
  const int qt = blockIdx.x;                    // 0..127 (uniform per block)
  const int bh = blockIdx.y * 4 + widx;         // 0..39
  const int b = bh / 10, h = bh % 10;
  const int qbase = qt * 16;

  const unsigned short* Q  = qbuf + (size_t)bh * 2048 * 64;
  const unsigned short* Kp = kbuf + (size_t)bh * 2048 * 64;
  const unsigned short* Vt = vbuf + (size_t)bh * 64 * 2048;

  // Q fragments for d-chunks 0..31 and 32..63 (A layout)
  Frag qa[2];
#pragma unroll
  for (int c = 0; c < 2; ++c)
#pragma unroll
    for (int v = 0; v < 8; ++v)
      qa[c].d[v] = *(const unsigned int*)(Q + (size_t)(qbase + ln) * 64 +
                                          c * 32 + a_kpair(v, hi));

  float mi[8], li[8];
  v8f O[4] = {};
#pragma unroll
  for (int r = 0; r < 8; ++r) { mi[r] = -1e30f; li[r] = 0.0f; }

  const int jmax = qbase + 16;                  // causal extent
  for (int j = 0; j < jmax; j += 32) {
    // ---- scores S = (Q K^T) * 1/8 for 32 keys -------------------------
    v8f sc[2];
#pragma unroll
    for (int t = 0; t < 2; ++t) {
      const int krow = min(j + t * 16 + ln, 2047);
      Frag kb0, kb1;
#pragma unroll
      for (int v = 0; v < 8; ++v) {
        kb0.d[v] = *(const unsigned int*)(Kp + (size_t)krow * 64 + b_kpair(v, hi));
        kb1.d[v] = *(const unsigned int*)(Kp + (size_t)krow * 64 + 32 + b_kpair(v, hi));
      }
      v8f a = {};
      a = wmma_bf16(qa[0], kb0, a);
      a = wmma_bf16(qa[1], kb1, a);
      sc[t] = a;
    }
    // ---- causal mask + scale -----------------------------------------
#pragma unroll
    for (int r = 0; r < 8; ++r) {
      const int qi = qbase + r + 8 * hi;
#pragma unroll
      for (int t = 0; t < 2; ++t) {
        const int key = j + t * 16 + ln;
        const float s = sc[t][r] * 0.125f;
        sc[t][r] = (key <= qi) ? s : -1e30f;
      }
    }
    // ---- online softmax update ---------------------------------------
#pragma unroll
    for (int r = 0; r < 8; ++r) {
      float mx = fmaxf(sc[0][r], sc[1][r]);
#pragma unroll
      for (int off = 1; off <= 8; off <<= 1)
        mx = fmaxf(mx, __shfl_xor(mx, off, 32));   // row-wise (16-lane half)
      const float mnew = fmaxf(mi[r], mx);
      const float scale = __expf(mi[r] - mnew);
      mi[r] = mnew;
      const float p0 = __expf(sc[0][r] - mnew);
      const float p1 = __expf(sc[1][r] - mnew);
      float rs = p0 + p1;
#pragma unroll
      for (int off = 1; off <= 8; off <<= 1)
        rs += __shfl_xor(rs, off, 32);
      li[r] = li[r] * scale + rs;
#pragma unroll
      for (int t4 = 0; t4 < 4; ++t4) O[t4][r] *= scale;
      const int mrow = r + 8 * hi;                 // C-layout row
      pbuf[widx][mrow][ln]      = f32_to_bf16(p0);
      pbuf[widx][mrow][16 + ln] = f32_to_bf16(p1);
    }
    __syncthreads();                               // C-layout -> A-layout via LDS
    Frag pa;
#pragma unroll
    for (int v = 0; v < 8; ++v)
      pa.d[v] = *(const unsigned int*)&pbuf[widx][ln][a_kpair(v, hi)];
    __syncthreads();
    // ---- O += P * V (V^T stored [d][s], so pairs along k are dwords) --
#pragma unroll
    for (int t4 = 0; t4 < 4; ++t4) {
      Frag vb;
#pragma unroll
      for (int v = 0; v < 8; ++v) {
        const int sidx = min(j + b_kpair(v, hi), 2046);
        vb.d[v] = *(const unsigned int*)(Vt + (size_t)(t4 * 16 + ln) * 2048 + sidx);
      }
      O[t4] = wmma_bf16(pa, vb, O[t4]);
    }
  }

  // ---- epilogue: y = O / l, scatter to [B*S, 640] bf16 -----------------
#pragma unroll
  for (int t4 = 0; t4 < 4; ++t4)
#pragma unroll
    for (int r = 0; r < 8; ++r) {
      const int mrow = r + 8 * hi;
      const size_t row = (size_t)b * 2048 + qbase + mrow;
      const int col = h * 64 + t4 * 16 + ln;
      yb[row * 640 + col] = f32_to_bf16(O[t4][r] / li[r]);
    }
}

// ---------------------------------------------------------------------------
// K3: out = y @ w_proj^T (fp32 out). 128 x 10 macro-tiles (64x64).
// ---------------------------------------------------------------------------
__global__ __launch_bounds__(128)
void proj_gemm_kernel(const unsigned short* __restrict__ yb,
                      const unsigned short* __restrict__ wb,
                      float* __restrict__ out) {
  const int lane = threadIdx.x & 31;
  const int hi = lane >> 4, ln = lane & 15;
  const int tile = blockIdx.x * 4 + (threadIdx.x >> 5);
  const int mt = tile / 10, nt = tile % 10;
  const int mbase = mt * 64, nbase = nt * 64;

  v8f acc[4][4] = {};
  gemm_64x64_k640(yb, wb, mbase, nbase, lane, acc);

#pragma unroll
  for (int mi = 0; mi < 4; ++mi)
#pragma unroll
    for (int ni = 0; ni < 4; ++ni)
#pragma unroll
      for (int r = 0; r < 8; ++r) {
        const int m = mi * 16 + r + 8 * hi;
        out[(size_t)(mbase + m) * 640 + nbase + ni * 16 + ln] = acc[mi][ni][r];
      }
}

// ---------------------------------------------------------------------------
// Launch
// ---------------------------------------------------------------------------
extern "C" void kernel_launch(void* const* d_in, const int* in_sizes, int n_in,
                              void* d_out, int out_size, void* d_ws, size_t ws_size,
                              hipStream_t stream) {
  (void)in_sizes; (void)n_in; (void)out_size; (void)ws_size;
  const float* x      = (const float*)d_in[0];   // [4,2048,640]
  const float* w_qkv  = (const float*)d_in[1];   // [1920,640]
  const float* w_proj = (const float*)d_in[2];   // [640,640]
  float* out = (float*)d_out;                    // [4,2048,640]

  const size_t XB_N   = 8192ull * 640;           // 5,242,880
  const size_t WQKV_N = 1920ull * 640;           // 1,228,800
  const size_t WPRJ_N = 640ull  * 640;           //   409,600
  const size_t QKV_N  = 40ull * 2048 * 64;       // 5,242,880

  unsigned short* ws    = (unsigned short*)d_ws;
  unsigned short* xb    = ws;
  unsigned short* wqkvb = xb    + XB_N;
  unsigned short* wprjb = wqkvb + WQKV_N;
  unsigned short* qb    = wprjb + WPRJ_N;
  unsigned short* kb    = qb    + QKV_N;
  unsigned short* vb    = kb    + QKV_N;
  unsigned short* ybuf  = vb    + QKV_N;

  cvt_bf16_kernel<<<(int)(XB_N   / 256), 256, 0, stream>>>(x,      xb,    (int)XB_N);
  cvt_bf16_kernel<<<(int)(WQKV_N / 256), 256, 0, stream>>>(w_qkv,  wqkvb, (int)WQKV_N);
  cvt_bf16_kernel<<<(int)(WPRJ_N / 256), 256, 0, stream>>>(w_proj, wprjb, (int)WPRJ_N);

  // 128*30 macro-tiles / 4 waves per block
  qkv_gemm_kernel<<<960, 128, 0, stream>>>(xb, wqkvb, qb, kb, vb);

  // 128 q-tiles x 10 groups of 4 (b,h) pairs
  attn_kernel<<<dim3(128, 10), 128, 0, stream>>>(qb, kb, vb, ybuf);

  // 128*10 macro-tiles / 4 waves per block
  proj_gemm_kernel<<<320, 128, 0, stream>>>(ybuf, wprjb, out);
}
